// MultiHeadAttention_32177894981871
// MI455X (gfx1250) — compile-verified, matrix operand order checked by reading, not run on hardware
//
#include <hip/hip_runtime.h>

typedef __attribute__((ext_vector_type(16))) _Float16 v16h;
typedef __attribute__((ext_vector_type(8)))  float    v8f;

#define NH    16
#define HD    64
#define T_CTX 2048
#define EMB   1024
#define NB    2
#define MTOT  (NB * T_CTX)   // 4096 tokens total

// ---------------------------------------------------------------------------
// WMMA helper: D = A(16x32 f16) * B(32x16 f16) + C(16x16 f32)
// ---------------------------------------------------------------------------
__device__ __forceinline__ v8f wmma_f16(v16h a, v16h b, v8f c) {
  return __builtin_amdgcn_wmma_f32_16x16x32_f16(
      /*neg_a=*/false, a, /*neg_b=*/false, b,
      /*c_mod=*/(short)0, c, /*reuse_a=*/false, /*reuse_b=*/false);
}

// ---------------------------------------------------------------------------
// Load a 16x32 fragment whose per-lane dimension is a contiguous row of
// memory (stride `ld` halves). Used for A operands (rows = M along K) and
// for B operands when the source is stored transposed (rows = N along K).
// CDNA5 layout: lanes 0-15 hold K {0..7, 16..23}, lanes 16-31 {8..15, 24..31}.
// Two 128-bit loads per lane.
// ---------------------------------------------------------------------------
__device__ __forceinline__ v16h load_frag_rows(const _Float16* __restrict__ base, int ld) {
  const int lane = threadIdx.x & 31;
  const int r    = lane & 15;
  const int ko   = (lane >> 4) << 3;          // 0 or 8
  const _Float16* p = base + (size_t)r * ld + ko;
  union { v16h h; uint4 u[2]; } f;
  f.u[0] = *(const uint4*)(p);                // K = ko .. ko+7
  f.u[1] = *(const uint4*)(p + 16);           // K = ko+16 .. ko+23
  return f.h;
}

// ---------------------------------------------------------------------------
// fp32 -> fp16 conversion (linear)
// ---------------------------------------------------------------------------
__global__ void cvt_f32_to_f16(const float* __restrict__ in,
                               _Float16* __restrict__ out, int n) {
  int i = blockIdx.x * blockDim.x + threadIdx.x;
  if (i < n) out[i] = (_Float16)in[i];
}

// ---------------------------------------------------------------------------
// Fused transpose + fp32 -> fp16 for EMB x EMB weights (LDS tiled).
// out[n][k] = (f16) in[k][n]
// ---------------------------------------------------------------------------
__global__ __launch_bounds__(256) void cvt_transpose_f16(
    const float* __restrict__ in, _Float16* __restrict__ out) {
  __shared__ float tile[16][17];
  const int bx = blockIdx.x * 16;   // n base
  const int by = blockIdx.y * 16;   // k base
  const int tx = threadIdx.x;       // 0..15
  const int ty = threadIdx.y;       // 0..15
  tile[ty][tx] = in[(size_t)(by + ty) * EMB + (bx + tx)];
  __syncthreads();
  out[(size_t)(bx + ty) * EMB + (by + tx)] = (_Float16)tile[tx][ty];
}

// ---------------------------------------------------------------------------
// QKV projection: out = X[4096,1024] * W + bias, W given TRANSPOSED (WT[n][k]).
// One wave = 64(M) x 32(N) tile, K in steps of 32: 8 WMMAs per step.
// layout 0: out[b,h,t,d]  (Q, K)        layout 1: out[b,h,d,t]  (V transposed)
// ---------------------------------------------------------------------------
__global__ __launch_bounds__(256) void qkv_gemm(
    const _Float16* __restrict__ X,      // [4096, 1024]
    const _Float16* __restrict__ WT,     // [1024, 1024] transposed
    const float*    __restrict__ bias,   // [1024]
    _Float16*       __restrict__ out,
    int layout) {
  const int wave  = (int)((blockIdx.x * blockDim.x + threadIdx.x) >> 5);
  const int ntile = wave & 31;           // EMB/32 = 32
  const int mtile = wave >> 5;           // MTOT/64 = 64
  const int m0 = mtile * 64;
  const int n0 = ntile * 32;

  v8f acc[2][4] = {};
  for (int k = 0; k < EMB; k += 32) {
    v16h b0 = load_frag_rows(WT + (size_t)n0 * EMB + k, EMB);
    v16h b1 = load_frag_rows(WT + (size_t)(n0 + 16) * EMB + k, EMB);
#pragma unroll
    for (int t = 0; t < 4; ++t) {
      v16h a = load_frag_rows(X + (size_t)(m0 + 16 * t) * EMB + k, EMB);
      acc[0][t] = wmma_f16(a, b0, acc[0][t]);
      acc[1][t] = wmma_f16(a, b1, acc[1][t]);
    }
  }

  const int lane = threadIdx.x & 31;
  const int rb   = (lane >> 4) << 3;     // 0 or 8 (row offset within subtile)
#pragma unroll
  for (int u = 0; u < 2; ++u) {
    const int n = n0 + 16 * u + (lane & 15);
    const int h = n >> 6;
    const int d = n & 63;
    const float bv = bias[n];
#pragma unroll
    for (int t = 0; t < 4; ++t) {
      const int mbase = m0 + 16 * t + rb;        // 8 consecutive rows
      const int bi    = mbase >> 11;             // / T_CTX (subtile in one batch)
      const int tt0   = mbase & (T_CTX - 1);
      if (layout == 0) {
        // [b,h,t,d]: column-strided scalar stores
#pragma unroll
        for (int r = 0; r < 8; ++r)
          out[(((size_t)bi * NH + h) * T_CTX + (tt0 + r)) * HD + d] =
              (_Float16)(acc[u][t][r] + bv);
      } else {
        // [b,h,d,t]: 8 consecutive t -> one 128-bit store
        union { uint4 q; _Float16 hh[8]; } pk;
#pragma unroll
        for (int r = 0; r < 8; ++r) pk.hh[r] = (_Float16)(acc[u][t][r] + bv);
        *(uint4*)(out + (((size_t)bi * NH + h) * HD + d) * T_CTX + tt0) = pk.q;
      }
    }
  }
}

// ---------------------------------------------------------------------------
// Flash attention, one wave per (b, h, 16-query block).
// S^T = K_blk(32x64) * Q^T(64x16): softmax result lands directly in the
// B-operand layout of P^T (in-lane f32->f16 packing only). Then
// O^T += V^T * P^T with V stored pre-transposed as [B,H,D,T] so the V^T
// A-fragments are plain 128-bit row loads. Online softmax, causal mask.
// ---------------------------------------------------------------------------
__global__ __launch_bounds__(256) void flash_attn(
    const _Float16* __restrict__ Q,    // [NB, NH, T_CTX, HD]
    const _Float16* __restrict__ K,    // [NB, NH, T_CTX, HD]
    const _Float16* __restrict__ VT,   // [NB, NH, HD, T_CTX]  (transposed)
    _Float16*       __restrict__ O) {  // [NB, T_CTX, NH*HD]
  const int wave = (int)((blockIdx.x * blockDim.x + threadIdx.x) >> 5);
  const int qblk = wave & 127;          // T_CTX/16 = 128
  const int bh   = wave >> 7;           // 0..31
  const int b    = bh >> 4;
  const int h    = bh & 15;

  const _Float16* Qp  = Q  + (size_t)bh * T_CTX * HD;
  const _Float16* Kp  = K  + (size_t)bh * T_CTX * HD;
  const _Float16* VTp = VT + (size_t)bh * HD * T_CTX;

  const int lane  = threadIdx.x & 31;
  const int klane = (lane >> 4) << 3;   // 0 or 8
  const int q0    = qblk * 16;
  const int qg    = q0 + (lane & 15);   // this lane's query (columns of S^T)
  const float scale = 0.03125f;         // 1/sqrt(EMB) per reference

  // Q as B operand of S^T = K * Q^T : rows of Q along d.
  v16h qb0 = load_frag_rows(Qp + (size_t)q0 * HD + 0,  HD);
  v16h qb1 = load_frag_rows(Qp + (size_t)q0 * HD + 32, HD);

  v8f  o_acc[4] = {};
  float m_run = -1e30f, l_run = 0.0f;

  const int jmax = (q0 + 15) >> 5;      // inclusive; 32-key blocks
  for (int j = 0; j <= jmax; ++j) {
    const int k0 = j * 32;

    // S^T tiles: tile t covers keys k0+16t..+15 (rows), queries (cols)
    v8f st[2];
#pragma unroll
    for (int t = 0; t < 2; ++t) {
      v16h ka0 = load_frag_rows(Kp + (size_t)(k0 + 16 * t) * HD + 0,  HD);
      v16h ka1 = load_frag_rows(Kp + (size_t)(k0 + 16 * t) * HD + 32, HD);
      v8f c = {};
      c = wmma_f16(ka0, qb0, c);
      c = wmma_f16(ka1, qb1, c);
      st[t] = c;
    }

    // scale + causal mask (key index kg = k0 + 16t + klane + r)
    const bool need_mask = (k0 + 31 > q0);
    float s[16];
#pragma unroll
    for (int t = 0; t < 2; ++t) {
#pragma unroll
      for (int r = 0; r < 8; ++r) {
        float val = st[t][r] * scale;
        int kg = k0 + 16 * t + klane + r;
        if (need_mask && kg > qg) val = -3.0e38f;
        s[t * 8 + r] = val;
      }
    }

    // online softmax stats; lanes L and L+16 hold the same query column
    float bm = s[0];
#pragma unroll
    for (int i = 1; i < 16; ++i) bm = fmaxf(bm, s[i]);
    bm = fmaxf(bm, __shfl_xor(bm, 16, 32));
    const float m_new = fmaxf(m_run, bm);
    const float corr  = __expf(m_run - m_new);

    v16h pb;                            // P^T directly in B-operand layout
    float ls = 0.0f;
#pragma unroll
    for (int i = 0; i < 16; ++i) {
      float e = __expf(s[i] - m_new);
      ls += e;
      pb[i] = (_Float16)e;
    }
    ls += __shfl_xor(ls, 16, 32);
    l_run = l_run * corr + ls;
    m_run = m_new;

    // rescale running output, then O^T += V^T * P^T (4 d-tiles of 16)
#pragma unroll
    for (int t = 0; t < 4; ++t) {
#pragma unroll
      for (int r = 0; r < 8; ++r) o_acc[t][r] *= corr;
    }
#pragma unroll
    for (int t = 0; t < 4; ++t) {
      v16h va = load_frag_rows(VTp + (size_t)(16 * t) * T_CTX + k0, T_CTX);
      o_acc[t] = wmma_f16(va, pb, o_acc[t]);
    }
  }

  // epilogue: O[q, h*64 + d] = o_acc / l ; lane stores 8 contiguous halves
  const float inv_l = 1.0f / l_run;
  const size_t orow = ((size_t)b * T_CTX + qg) * (NH * HD) + (size_t)h * HD;
#pragma unroll
  for (int t = 0; t < 4; ++t) {
    union { uint4 u; _Float16 hh[8]; } pk;
#pragma unroll
    for (int r = 0; r < 8; ++r) pk.hh[r] = (_Float16)(o_acc[t][r] * inv_l);
    *(uint4*)(O + orow + 16 * t + klane) = pk.u;
  }
}

// ---------------------------------------------------------------------------
// Output projection: out[m,n] = sum_k A[m,k] * Wo[k,n] + bo[n] (fp32 out).
// Wo given transposed. Wave = 64x32 tile, 8 WMMAs per 32-deep k-step.
// ---------------------------------------------------------------------------
__global__ __launch_bounds__(256) void out_gemm(
    const _Float16* __restrict__ A,      // [4096, 1024]
    const _Float16* __restrict__ WT,     // [1024, 1024] transposed
    const float*    __restrict__ bias,   // [1024]
    float*          __restrict__ out) {  // [4096, 1024] fp32
  const int wave  = (int)((blockIdx.x * blockDim.x + threadIdx.x) >> 5);
  const int ntile = wave & 31;
  const int mtile = wave >> 5;
  const int m0 = mtile * 64;
  const int n0 = ntile * 32;

  v8f acc[2][4] = {};
  for (int k = 0; k < EMB; k += 32) {
    v16h b0 = load_frag_rows(WT + (size_t)n0 * EMB + k, EMB);
    v16h b1 = load_frag_rows(WT + (size_t)(n0 + 16) * EMB + k, EMB);
#pragma unroll
    for (int t = 0; t < 4; ++t) {
      v16h a = load_frag_rows(A + (size_t)(m0 + 16 * t) * EMB + k, EMB);
      acc[0][t] = wmma_f16(a, b0, acc[0][t]);
      acc[1][t] = wmma_f16(a, b1, acc[1][t]);
    }
  }

  const int lane = threadIdx.x & 31;
  const int rb   = (lane >> 4) << 3;
#pragma unroll
  for (int u = 0; u < 2; ++u) {
    const int n = n0 + 16 * u + (lane & 15);
    const float bv = bias[n];
#pragma unroll
    for (int t = 0; t < 4; ++t) {
#pragma unroll
      for (int r = 0; r < 8; ++r) {
        int m = m0 + 16 * t + rb + r;
        out[(size_t)m * EMB + n] = acc[u][t][r] + bv;
      }
    }
  }
}

// ---------------------------------------------------------------------------
// Host launcher
// ---------------------------------------------------------------------------
extern "C" void kernel_launch(void* const* d_in, const int* in_sizes, int n_in,
                              void* d_out, int out_size, void* d_ws, size_t ws_size,
                              hipStream_t stream) {
  (void)in_sizes; (void)n_in; (void)out_size; (void)ws_size;
  const float* x  = (const float*)d_in[0];
  const float* Wk = (const float*)d_in[1];
  const float* bk = (const float*)d_in[2];
  const float* Wq = (const float*)d_in[3];
  const float* bq = (const float*)d_in[4];
  const float* Wv = (const float*)d_in[5];
  const float* bv = (const float*)d_in[6];
  const float* Wo = (const float*)d_in[7];
  const float* bo = (const float*)d_in[8];
  float* out = (float*)d_out;

  const size_t NX = (size_t)MTOT * EMB;      // 4,194,304
  const size_t NW = (size_t)EMB * EMB;       // 1,048,576

  _Float16* x_h    = (_Float16*)d_ws;
  _Float16* wq_t   = x_h  + NX;              // transposed f16 weights
  _Float16* wk_t   = wq_t + NW;
  _Float16* wv_t   = wk_t + NW;
  _Float16* wo_t   = wv_t + NW;
  _Float16* q_h    = wo_t + NW;              // [NB,NH,T,HD]
  _Float16* k_h    = q_h  + NX;              // [NB,NH,T,HD]
  _Float16* vt_h   = k_h  + NX;              // [NB,NH,HD,T]
  _Float16* attn_h = vt_h + NX;              // [NB,T,NH*HD]

  // Stage 0: conversions (x linear; weights fused transpose+convert)
  cvt_f32_to_f16<<<(int)((NX + 255) / 256), 256, 0, stream>>>(x, x_h, (int)NX);
  dim3 tgrid(EMB / 16, EMB / 16), tblk(16, 16);
  cvt_transpose_f16<<<tgrid, tblk, 0, stream>>>(Wq, wq_t);
  cvt_transpose_f16<<<tgrid, tblk, 0, stream>>>(Wk, wk_t);
  cvt_transpose_f16<<<tgrid, tblk, 0, stream>>>(Wv, wv_t);
  cvt_transpose_f16<<<tgrid, tblk, 0, stream>>>(Wo, wo_t);

  // Stage 1: QKV projections (64 x 32 tiles -> 2048 waves = 256 blocks x 8)
  qkv_gemm<<<256, 256, 0, stream>>>(x_h, wq_t, bq, q_h, 0);
  qkv_gemm<<<256, 256, 0, stream>>>(x_h, wk_t, bk, k_h, 0);
  qkv_gemm<<<256, 256, 0, stream>>>(x_h, wv_t, bv, vt_h, 1);

  // Stage 2: flash attention (NB*NH*(T/16) = 4096 waves = 512 blocks x 8)
  flash_attn<<<512, 256, 0, stream>>>(q_h, k_h, vt_h, attn_h);

  // Stage 3: output projection (fp32 result)
  out_gemm<<<256, 256, 0, stream>>>(attn_h, wo_t, bo, out);
}
